// ImitateJoint_87857851007056
// MI455X (gfx1250) — matrix operand order repaired and checked by reading, not compiled, not run on hardware
//
#include <hip/hip_runtime.h>
#include <math.h>

#define DEVINL __device__ __forceinline__

typedef __bf16 bf16;
typedef __attribute__((ext_vector_type(8)))  __bf16 v8bf;
typedef __attribute__((ext_vector_type(16))) __bf16 v16bf;
typedef __attribute__((ext_vector_type(8)))  float  v8f;

// Problem constants (match reference)
constexpr int HD   = 1024;
constexpr int INSZ = 1024;
constexpr int NV   = 400;     // vocab V
constexpr int OPSZ = 128;
constexpr int BN   = 64;      // batch
constexpr int KB   = 8;       // beams
constexpr int LP   = 13;      // program_len
constexpr int RR   = BN * KB; // 512 rows
constexpr int NG   = 4 * HD;  // 4096 gate columns
constexpr float NEGBIG = -3.402823e38f;

// ---------------------------------------------------------------------------
// helpers
// ---------------------------------------------------------------------------
DEVINL float sigm(float x) { return 1.f / (1.f + __expf(-x)); }

DEVINL float log1mexp_f(float x) {
  // log(1 - exp(x)) for x <= 0, stable on both ends (matches reference)
  x = fminf(x, -1e-38f);
  if (x > -0.6931472f) return logf(-expm1f(x));
  return log1pf(-expf(x));
}

DEVINL float wave_max_f(float v) {
#pragma unroll
  for (int o = 16; o > 0; o >>= 1) v = fmaxf(v, __shfl_xor(v, o, 32));
  return v;
}
DEVINL float wave_sum_f(float v) {
#pragma unroll
  for (int o = 16; o > 0; o >>= 1) v += __shfl_xor(v, o, 32);
  return v;
}

// ---------------------------------------------------------------------------
// WMMA fragment loads (bf16, 16x16x32), layouts per CDNA5 ISA 7.12.2
// ---------------------------------------------------------------------------
DEVINL v16bf frag_cat(v8bf lo, v8bf hi) {
  v16bf r;
#pragma unroll
  for (int i = 0; i < 8; ++i) { r[i] = lo[i]; r[i + 8] = hi[i]; }
  return r;
}

// A fragment: base pa = A + row*lda + (lane>>4)*8, element offset ke;
// halves at ke and ke+16.
DEVINL v16bf fragA_at(const bf16* __restrict__ pa, int ke) {
  return frag_cat(*(const v8bf*)(pa + ke), *(const v8bf*)(pa + ke + 16));
}
// B fragment: base pb = W + col*ldw + (lane>>4)*8, element offset ke;
// halves at ke and ke+16. Constant column displacements fold into the
// 24-bit instruction offset (<= 6 MB here).
DEVINL v16bf fragB_at(const bf16* __restrict__ pb, size_t ke) {
  return frag_cat(*(const v8bf*)(pb + ke), *(const v8bf*)(pb + ke + 16));
}

#define WMMA_BF16(A_, B_, C_) \
  __builtin_amdgcn_wmma_f32_16x16x32_bf16(false, (A_), false, (B_), (short)0, (C_), false, false)

// ---------------------------------------------------------------------------
// One K-pass over NCH chunks accumulating TWO 16x16 output tiles.
// Ping-pong double buffering with peeled 2-chunk epilogue. Live fragment set
// = aX/aY + 2x2 B tiles (~48 VGPRs) so the allocator never spills at the
// 128-VGPR budget while still overlapping VMEM with the matrix pipe.
// ---------------------------------------------------------------------------
template <int NCH>
DEVINL void kpass2(const bf16* __restrict__ pa, const bf16* __restrict__ pb,
                   size_t c0, size_t c1, v8f& a0, v8f& a1) {
  v16bf aX = fragA_at(pa, 0);
  v16bf b0X = fragB_at(pb, c0), b1X = fragB_at(pb, c1);
  v16bf aY, b0Y, b1Y;
  for (int ch = 0; ch < NCH - 2; ch += 2) {
    int k1 = (ch + 1) << 5;
    aY = fragA_at(pa, k1);
    b0Y = fragB_at(pb, c0 + k1);
    b1Y = fragB_at(pb, c1 + k1);
    __builtin_prefetch(pb + c0 + ((ch + 4) << 5), 0, 1);  // speculative
    a0 = WMMA_BF16(aX, b0X, a0);
    a1 = WMMA_BF16(aX, b1X, a1);
    int k2 = (ch + 2) << 5;
    aX = fragA_at(pa, k2);
    b0X = fragB_at(pb, c0 + k2);
    b1X = fragB_at(pb, c1 + k2);
    a0 = WMMA_BF16(aY, b0Y, a0);
    a1 = WMMA_BF16(aY, b1Y, a1);
  }
  constexpr int kl = (NCH - 1) * 32;
  aY = fragA_at(pa, kl);
  b0Y = fragB_at(pb, c0 + kl);
  b1Y = fragB_at(pb, c1 + kl);
  a0 = WMMA_BF16(aX, b0X, a0);
  a1 = WMMA_BF16(aX, b1X, a1);
  a0 = WMMA_BF16(aY, b0Y, a0);
  a1 = WMMA_BF16(aY, b1Y, a1);
}

template <int NCH>
DEVINL void kpass1(const bf16* __restrict__ pa, const bf16* __restrict__ pb,
                   size_t c0, v8f& a0) {
  v16bf aX = fragA_at(pa, 0);
  v16bf b0X = fragB_at(pb, c0);
  v16bf aY, b0Y;
  for (int ch = 0; ch < NCH - 2; ch += 2) {
    int k1 = (ch + 1) << 5;
    aY = fragA_at(pa, k1);
    b0Y = fragB_at(pb, c0 + k1);
    a0 = WMMA_BF16(aX, b0X, a0);
    int k2 = (ch + 2) << 5;
    aX = fragA_at(pa, k2);
    b0X = fragB_at(pb, c0 + k2);
    a0 = WMMA_BF16(aY, b0Y, a0);
  }
  constexpr int kl = (NCH - 1) * 32;
  aY = fragA_at(pa, kl);
  b0Y = fragB_at(pb, c0 + kl);
  a0 = WMMA_BF16(aX, b0X, a0);
  a0 = WMMA_BF16(aY, b0Y, a0);
}

// ---------------------------------------------------------------------------
// Generic GEMM: C[M x N] = A[M x K](bf16) * W[N x K](bf16)^T + bias
// wave tile = 16 x (16*NT), computed as NT/2 paired K-passes.
// ---------------------------------------------------------------------------
template <int NT, int LDA, int LDW, int KD, bool RELU, bool STF32, bool STBF>
__global__ __attribute__((amdgpu_waves_per_eu(1)))
void gemm_bf16_kernel(const bf16* __restrict__ A, const bf16* __restrict__ W,
                      const float* __restrict__ bias,
                      float* __restrict__ Cf, bf16* __restrict__ Cb,
                      int Mtiles, int N) {
  int wave = (blockIdx.x * blockDim.x + threadIdx.x) >> 5;
  int lane = threadIdx.x & 31;
  int ngroups = N / (16 * NT);
  if (wave >= Mtiles * ngroups) return;     // wave-uniform exit
  int mt = wave / ngroups, ng = wave % ngroups;
  int m0 = mt * 16, n0 = ng * 16 * NT;

  const bf16* pa = A + (size_t)(m0 + (lane & 15)) * LDA + ((lane >> 4) << 3);
  const bf16* pb = W + (size_t)(n0 + (lane & 15)) * LDW + ((lane >> 4) << 3);

  v8f acc[NT];
#pragma unroll
  for (int j = 0; j < NT; ++j)
#pragma unroll
    for (int v = 0; v < 8; ++v) acc[j][v] = 0.f;

#pragma unroll
  for (int j0 = 0; j0 + 1 < NT; j0 += 2)
    kpass2<KD / 32>(pa, pb, (size_t)j0 * 16 * LDW, (size_t)(j0 + 1) * 16 * LDW,
                    acc[j0], acc[j0 + 1]);
  if (NT & 1)
    kpass1<KD / 32>(pa, pb, (size_t)(NT - 1) * 16 * LDW, acc[NT - 1]);

  int hi = lane >> 4, ln = lane & 15;
#pragma unroll
  for (int j = 0; j < NT; ++j) {
    int n = n0 + 16 * j + ln;
    float bv = bias ? bias[n] : 0.f;
#pragma unroll
    for (int v = 0; v < 8; ++v) {
      int m = m0 + v + 8 * hi;
      float val = acc[j][v] + bv;
      if (RELU) val = fmaxf(val, 0.f);
      if (STF32) Cf[(size_t)m * N + n] = val;
      if (STBF) Cb[(size_t)m * N + n] = (bf16)val;
    }
  }
}

// ---------------------------------------------------------------------------
// Fused gates GEMM + LSTM cell.
// gates = xc[r%64] + table[sample_id[r]] + h @ Whh^T   (table holds b_ih+b_hh)
// wave owns one 16x16 h-tile; 4 accumulators = i,f,g,o gate tiles whose B
// bases differ by the compile-time constant g*HD*HD (2/4/6 MB -> IOFFSET).
// Two paired K-passes: gates {i,f} then {g,o}.
// grid = 32 Mtiles * 64 Ntiles waves = 256 blocks of 256 threads.
// ---------------------------------------------------------------------------
__global__ __attribute__((amdgpu_waves_per_eu(1)))
void gates_lstm_kernel(const bf16* __restrict__ h, const bf16* __restrict__ Whh,
                       const float* __restrict__ xc, const float* __restrict__ table,
                       const int* __restrict__ sid,
                       const float* __restrict__ c_in, float* __restrict__ c_out,
                       bf16* __restrict__ h_out) {
  int wave = (blockIdx.x * blockDim.x + threadIdx.x) >> 5;
  int lane = threadIdx.x & 31;
  int mt = wave >> 6, nt = wave & 63;
  int m0 = mt * 16, n0 = nt * 16;
  int hi = lane >> 4, ln = lane & 15;
  constexpr size_t GSTRIDE = (size_t)HD * HD;  // B displacement per gate

  int svec[8];
#pragma unroll
  for (int v = 0; v < 8; ++v) svec[v] = sid[m0 + v + 8 * hi];

  v8f acc[4];
#pragma unroll
  for (int g = 0; g < 4; ++g) {
    int n = n0 + g * HD + ln;
#pragma unroll
    for (int v = 0; v < 8; ++v) {
      int m = m0 + v + 8 * hi;
      acc[g][v] = xc[(size_t)(m & 63) * NG + n] + table[(size_t)svec[v] * NG + n];
    }
  }

  const bf16* pa = h + (size_t)(m0 + (lane & 15)) * HD + ((lane >> 4) << 3);
  const bf16* pb = Whh + (size_t)(n0 + (lane & 15)) * HD + ((lane >> 4) << 3);

  kpass2<HD / 32>(pa, pb, 0, GSTRIDE, acc[0], acc[1]);
  kpass2<HD / 32>(pa, pb, 2 * GSTRIDE, 3 * GSTRIDE, acc[2], acc[3]);

  int n = n0 + ln;
#pragma unroll
  for (int v = 0; v < 8; ++v) {
    int m = m0 + v + 8 * hi;
    float ig = acc[0][v], fg = acc[1][v], gg = acc[2][v], og = acc[3][v];
    float c = sigm(fg) * c_in[(size_t)m * HD + n] + sigm(ig) * tanhf(gg);
    float hh = sigm(og) * tanhf(c);
    c_out[(size_t)m * HD + n] = c;
    h_out[(size_t)m * HD + n] = (bf16)hh;
  }
}

// ---------------------------------------------------------------------------
// Per-row: log_softmax, entropy, Gumbel-with-maximum. 512 blocks x 128 thr.
// ---------------------------------------------------------------------------
DEVINL float blk_max128(float v, float* sh4) {
  v = wave_max_f(v);
  if ((threadIdx.x & 31) == 0) sh4[threadIdx.x >> 5] = v;
  __syncthreads();
  float r = fmaxf(fmaxf(sh4[0], sh4[1]), fmaxf(sh4[2], sh4[3]));
  __syncthreads();
  return r;
}
DEVINL float blk_sum128(float v, float* sh4) {
  v = wave_sum_f(v);
  if ((threadIdx.x & 31) == 0) sh4[threadIdx.x >> 5] = v;
  __syncthreads();
  float r = sh4[0] + sh4[1] + sh4[2] + sh4[3];
  __syncthreads();
  return r;
}

__global__ void row_softmax_gumbel(const float* __restrict__ logits,
                                   const float* __restrict__ logp,
                                   const float* __restrict__ G,
                                   const float* __restrict__ gu_t,
                                   float* __restrict__ log_probs,
                                   float* __restrict__ gout,
                                   float* __restrict__ ent_pre) {
  __shared__ float sh4[4];
  int r = blockIdx.x, tid = threadIdx.x;
  float lg[4], gp[4];
  float lmax = NEGBIG;
#pragma unroll
  for (int i = 0; i < 4; ++i) {
    int v = tid + i * 128;
    lg[i] = (v < NV) ? logits[(size_t)r * NV + v] : NEGBIG;
    lmax = fmaxf(lmax, lg[i]);
  }
  lmax = blk_max128(lmax, sh4);
  float se = 0.f;
#pragma unroll
  for (int i = 0; i < 4; ++i) {
    int v = tid + i * 128;
    if (v < NV) se += __expf(lg[i] - lmax);
  }
  se = blk_sum128(se, sh4);
  float lse = lmax + logf(se);

  float lp_r = logp[r], G_r = G[r];
  float entacc = 0.f, zmax = NEGBIG;
#pragma unroll
  for (int i = 0; i < 4; ++i) {
    int v = tid + i * 128;
    if (v < NV) {
      float ls = lg[i] - lse;
      log_probs[(size_t)r * NV + v] = ls;
      entacc += ls * __expf(ls);
      float u = gu_t[(size_t)r * NV + v];
      float gmb = -logf(-logf(u * (1.f - 2e-7f) + 1e-7f));
      gp[i] = ls + lp_r + gmb;
      zmax = fmaxf(zmax, gp[i]);
    } else gp[i] = NEGBIG;
  }
  entacc = blk_sum128(entacc, sh4);
  if (tid == 0) ent_pre[r] = entacc;
  zmax = blk_max128(zmax, sh4);
#pragma unroll
  for (int i = 0; i < 4; ++i) {
    int v = tid + i * 128;
    if (v < NV) {
      float vv = G_r - gp[i] + log1mexp_f(gp[i] - zmax);
      gout[(size_t)r * NV + v] = G_r - fmaxf(vv, 0.f) - log1pf(__expf(-fabsf(vv)));
    }
  }
}

// ---------------------------------------------------------------------------
// Per-batch top-k selection + in-place state reorder (beams live in one WG).
// 64 blocks x 256 threads.
// ---------------------------------------------------------------------------
__global__ void select_topk(const float* __restrict__ g, const float* __restrict__ lpmat,
                            float* logp, float* G, float* ent, const float* __restrict__ ent_pre,
                            float* samples, float* outputs,
                            int* sid, int* order_row, float* phi_k, int t) {
  __shared__ float gbuf[KB * NV];
  __shared__ float rv[256];
  __shared__ int ri[256];
  __shared__ float selv[KB];
  __shared__ int seli[KB];
  __shared__ float o_logp[KB], o_ent[KB], o_samp[KB * LP], o_out[KB * LP];
  int b = blockIdx.x, tid = threadIdx.x;
  int nub = (t == 0) ? 1 : KB;

  for (int i = tid; i < KB * NV; i += 256) {
    int j = i / NV, v = i - j * NV;
    gbuf[i] = (j < nub) ? g[(size_t)(j * BN + b) * NV + v] : NEGBIG;
  }
  for (int i = tid; i < KB * LP; i += 256) {
    int j = i / LP, c = i - j * LP;
    o_samp[i] = samples[(size_t)(j * BN + b) * LP + c];
    o_out[i] = outputs[(size_t)(j * BN + b) * LP + c];
  }
  if (tid < KB) { o_logp[tid] = logp[tid * BN + b]; o_ent[tid] = ent_pre[tid * BN + b]; }
  __syncthreads();

  for (int it = 0; it < KB; ++it) {
    float best = NEGBIG;
    int bi = KB * NV;
    for (int i = tid; i < KB * NV; i += 256) {
      float val = gbuf[i];
      if (val > best || (val == best && i < bi)) { best = val; bi = i; }
    }
    rv[tid] = best; ri[tid] = bi;
    __syncthreads();
    for (int s = 128; s > 0; s >>= 1) {
      if (tid < s) {
        if (rv[tid + s] > rv[tid] || (rv[tid + s] == rv[tid] && ri[tid + s] < ri[tid])) {
          rv[tid] = rv[tid + s]; ri[tid] = ri[tid + s];
        }
      }
      __syncthreads();
    }
    if (tid == 0) { selv[it] = rv[0]; seli[it] = ri[0]; gbuf[ri[0]] = NEGBIG; }
    __syncthreads();
  }

  if (tid < KB) {
    int jj = tid, idx = seli[jj], pj = idx / NV, sm = idx - pj * NV;
    int r = jj * BN + b, pr = pj * BN + b;
    float lps = lpmat[(size_t)pr * NV + sm];
    float nl = o_logp[pj] + lps;
    logp[r] = nl;
    G[r] = selv[jj];
    ent[r] = o_ent[pj];
    sid[r] = sm;
    order_row[r] = pr;
    phi_k[b * KB + jj] = nl;
    for (int c = 0; c < LP; ++c) {
      samples[(size_t)r * LP + c] = (c == t) ? (float)sm : o_samp[pj * LP + c];
      outputs[(size_t)r * LP + c] = (c == t) ? lps : o_out[pj * LP + c];
    }
  }
}

// h/c gather by parent row (tmp -> cur). 2048 blocks x 256 threads.
__global__ void reorder_hc(const bf16* __restrict__ h_src, const float* __restrict__ c_src,
                           const int* __restrict__ order_row,
                           bf16* __restrict__ h_dst, float* __restrict__ c_dst) {
  int idx = blockIdx.x * blockDim.x + threadIdx.x;
  int r = idx >> 10, e = idx & 1023;
  int pr = order_row[r];
  h_dst[idx] = h_src[(size_t)pr * HD + e];
  c_dst[idx] = c_src[(size_t)pr * HD + e];
}

// ---------------------------------------------------------------------------
// compute_log_R phase A: T (64x1000x8) and base (64x1000). 250 blocks x 256.
// ---------------------------------------------------------------------------
__global__ void logR_points(const float* __restrict__ phi_k,
                            float* __restrict__ Tbuf, float* __restrict__ basebuf) {
  int idx = blockIdx.x * blockDim.x + threadIdx.x;
  if (idx >= BN * 1000) return;
  int b = idx / 1000, pt = idx - b * 1000;
  float p[KB], pS = 0.f;
#pragma unroll
  for (int j = 0; j < KB; ++j) { p[j] = __expf(phi_k[b * KB + j]); pS += p[j]; }
  float lu = logf(((float)pt + 0.5f) * 1e-3f);
  float bs = -pS * lu;
#pragma unroll
  for (int j = 0; j < KB; ++j) {
    float Tj = log1mexp_f(p[j] * lu);
    Tbuf[(size_t)idx * KB + j] = Tj;
    bs += Tj;
  }
  basebuf[idx] = bs;
}

// phase B: per-batch logsumexp reductions + entropy accumulation. 64 x 256.
__global__ void logR_reduce(const float* __restrict__ Tbuf, const float* __restrict__ basebuf,
                            const float* __restrict__ phi_k, const float* __restrict__ ent,
                            float* __restrict__ logRs, float* __restrict__ logRss,
                            float* __restrict__ ent_plot, float* __restrict__ entropy_el) {
  __shared__ float base_s[1000];
  __shared__ float T_s[8000];
  __shared__ float red[256];
  __shared__ float logIs_sh[KB], logIss_sh[KB * KB], logI_sh;
  int b = blockIdx.x, tid = threadIdx.x, lane = tid & 31, wid = tid >> 5;
  const float LOGN = 6.907755279f;  // log(1000)

  for (int i = tid; i < 1000; i += 256) base_s[i] = basebuf[b * 1000 + i];
  for (int i = tid; i < 8000; i += 256) T_s[i] = Tbuf[(size_t)b * 8000 + i];
  __syncthreads();

  // logI = logsumexp(base) - log N
  float m = NEGBIG;
  for (int i = tid; i < 1000; i += 256) m = fmaxf(m, base_s[i]);
  red[tid] = m; __syncthreads();
  for (int s = 128; s > 0; s >>= 1) { if (tid < s) red[tid] = fmaxf(red[tid], red[tid + s]); __syncthreads(); }
  m = red[0]; __syncthreads();
  float se = 0.f;
  for (int i = tid; i < 1000; i += 256) se += __expf(base_s[i] - m);
  red[tid] = se; __syncthreads();
  for (int s = 128; s > 0; s >>= 1) { if (tid < s) red[tid] += red[tid + s]; __syncthreads(); }
  if (tid == 0) logI_sh = m + logf(red[0]) - LOGN;
  __syncthreads();

  // logI_s[j]: wave j handles logsumexp(base - T_j)
  {
    int j = wid;
    float mj = NEGBIG;
    for (int n = lane; n < 1000; n += 32) mj = fmaxf(mj, base_s[n] - T_s[n * KB + j]);
    mj = wave_max_f(mj);
    float sj = 0.f;
    for (int n = lane; n < 1000; n += 32) sj += __expf(base_s[n] - T_s[n * KB + j] - mj);
    sj = wave_sum_f(sj);
    if (lane == 0) logIs_sh[j] = mj + logf(sj) - LOGN;
  }
  // logI_ss[j][l]: wave j handles pairs (j, 0..7)
  for (int q = 0; q < KB; ++q) {
    int j = wid, l = q;
    float mp = NEGBIG;
    for (int n = lane; n < 1000; n += 32)
      mp = fmaxf(mp, base_s[n] - T_s[n * KB + j] - T_s[n * KB + l]);
    mp = wave_max_f(mp);
    float sp = 0.f;
    for (int n = lane; n < 1000; n += 32)
      sp += __expf(base_s[n] - T_s[n * KB + j] - T_s[n * KB + l] - mp);
    sp = wave_sum_f(sp);
    if (lane == 0) logIss_sh[j * KB + l] = mp + logf(sp) - LOGN;
  }
  __syncthreads();

  if (tid < KB) logRs[b * KB + tid] = logIs_sh[tid] - logI_sh;
  if (tid < KB * KB) {
    int j = tid >> 3, l = tid & 7;
    logRss[b * KB * KB + tid] = (j == l) ? 0.f : (logIss_sh[tid] - logIs_sh[j]);
  }
  __syncthreads();
  if (tid == 0) {
    float Wsum = 0.f, wsum_e = 0.f;
    for (int j = 0; j < KB; ++j) {
      float w = __expf(phi_k[b * KB + j] + (logIs_sh[j] - logI_sh));
      Wsum += w;
      wsum_e += w * ent[j * BN + b];
    }
    ent_plot[b] += wsum_e;
    entropy_el[b] += wsum_e / Wsum;
  }
}

// ---------------------------------------------------------------------------
// precompute helpers
// ---------------------------------------------------------------------------
__global__ void f2b_kernel(const float* __restrict__ s, bf16* __restrict__ d, int n) {
  int i = blockIdx.x * blockDim.x + threadIdx.x;
  if (i < n) d[i] = (bf16)s[i];
}
__global__ void f2b_slice_kernel(const float* __restrict__ s, bf16* __restrict__ d,
                                 int rows, int cols, int sld, int c0) {
  int i = blockIdx.x * blockDim.x + threadIdx.x;
  if (i < rows * cols) {
    int r = i / cols, c = i - r * cols;
    d[i] = (bf16)s[(size_t)r * sld + c0 + c];
  }
}
// inp_vec[s,c] = relu(W_in_op[c,s] + b_in_op[c]); rows 401..415 zero-padded
__global__ void inpvec_kernel(const float* __restrict__ W, const float* __restrict__ bvec,
                              bf16* __restrict__ d) {
  int i = blockIdx.x * blockDim.x + threadIdx.x;
  if (i < 416 * OPSZ) {
    int s = i / OPSZ, c = i - s * OPSZ;
    float val = (s <= NV) ? fmaxf(W[(size_t)c * (NV + 1) + s] + bvec[c], 0.f) : 0.f;
    d[i] = (bf16)val;
  }
}
__global__ void bsum_kernel(const float* __restrict__ a, const float* __restrict__ b,
                            float* __restrict__ d) {
  int i = blockIdx.x * blockDim.x + threadIdx.x;
  if (i < NG) d[i] = a[i] + b[i];
}
__global__ void init_sid_kernel(int* sid) {
  int i = blockIdx.x * blockDim.x + threadIdx.x;
  if (i < RR) sid[i] = NV;  // initial temp = one_hot(V)
}

// ---------------------------------------------------------------------------
// final output packing (all f32)
// ---------------------------------------------------------------------------
__global__ void pack_out(const float* __restrict__ outputs, const float* __restrict__ samples,
                         const float* __restrict__ entropy_el, const float* __restrict__ logRs,
                         const float* __restrict__ logRss, const float* __restrict__ phi_k,
                         const float* __restrict__ ent_plot, float* __restrict__ out, int total) {
  int idx = blockIdx.x * blockDim.x + threadIdx.x;
  if (idx >= total) return;
  if (idx < 6656) {
    int i = idx / 104, c = idx - i * 104, j = c / LP, tt = c - j * LP;
    out[idx] = outputs[(size_t)(j * BN + i) * LP + tt];
  } else if (idx < 13312) {
    int x = idx - 6656;
    int i = x / 104, c = x - i * 104, j = c / LP, tt = c - j * LP;
    out[idx] = samples[(size_t)(j * BN + i) * LP + tt];
  } else if (idx < 13376) out[idx] = entropy_el[idx - 13312];
  else if (idx < 13888)  out[idx] = logRs[idx - 13376];
  else if (idx < 17984)  out[idx] = logRss[idx - 13888];
  else if (idx < 18496)  out[idx] = phi_k[idx - 17984];
  else                   out[idx] = ent_plot[idx - 18496];
}

// ---------------------------------------------------------------------------
// host driver
// ---------------------------------------------------------------------------
extern "C" void kernel_launch(void* const* d_in, const int* in_sizes, int n_in,
                              void* d_out, int out_size, void* d_ws, size_t ws_size,
                              hipStream_t stream) {
  (void)in_sizes; (void)n_in; (void)out_size; (void)ws_size;
  const float* x_feat = (const float*)d_in[0];
  const float* W_in_op = (const float*)d_in[1];
  const float* b_in_op = (const float*)d_in[2];
  const float* W_ih = (const float*)d_in[3];
  const float* W_hh = (const float*)d_in[4];
  const float* b_ih = (const float*)d_in[5];
  const float* b_hh = (const float*)d_in[6];
  const float* b_fc1 = (const float*)d_in[8];
  const float* W_out = (const float*)d_in[9];
  const float* b_out = (const float*)d_in[10];
  const float* gumbel = (const float*)d_in[11];

  char* ws = (char*)d_ws;
  size_t off = 0;
  auto alloc = [&](size_t bytes) -> char* {
    char* p = ws + off;
    off = (off + bytes + 255) & ~(size_t)255;
    return p;
  };
  bf16* Whh_b   = (bf16*)alloc((size_t)NG * HD * 2);
  bf16* Wfc1_b  = (bf16*)alloc((size_t)HD * HD * 2);
  bf16* Wout_b  = (bf16*)alloc((size_t)NV * HD * 2);
  bf16* Wih0_b  = (bf16*)alloc((size_t)NG * INSZ * 2);
  bf16* Wihop_b = (bf16*)alloc((size_t)NG * OPSZ * 2);
  bf16* xfeat_b = (bf16*)alloc((size_t)BN * INSZ * 2);
  bf16* inpvec_b= (bf16*)alloc((size_t)416 * OPSZ * 2);
  float* xc64   = (float*)alloc((size_t)BN * NG * 4);
  float* table  = (float*)alloc((size_t)416 * NG * 4);
  float* bsum   = (float*)alloc((size_t)NG * 4);
  bf16* h_cur   = (bf16*)alloc((size_t)RR * HD * 2);
  bf16* h_tmp   = (bf16*)alloc((size_t)RR * HD * 2);
  float* c_cur  = (float*)alloc((size_t)RR * HD * 4);
  float* c_tmp  = (float*)alloc((size_t)RR * HD * 4);
  bf16* hd_b    = (bf16*)alloc((size_t)RR * HD * 2);
  float* logits = (float*)alloc((size_t)RR * NV * 4);
  float* lpmat  = (float*)alloc((size_t)RR * NV * 4);
  float* gmat   = (float*)alloc((size_t)RR * NV * 4);
  float* Tbuf   = (float*)alloc((size_t)BN * 1000 * KB * 4);
  float* basebuf= (float*)alloc((size_t)BN * 1000 * 4);
  float* samples= (float*)alloc((size_t)RR * LP * 4);
  float* outputs= (float*)alloc((size_t)RR * LP * 4);
  float* logp   = (float*)alloc(RR * 4);
  float* Gg     = (float*)alloc(RR * 4);
  float* ent    = (float*)alloc(RR * 4);
  float* ent_pre= (float*)alloc(RR * 4);
  int*  sid     = (int*)alloc(RR * 4);
  int*  order_r = (int*)alloc(RR * 4);
  float* phi_k  = (float*)alloc(BN * KB * 4);
  float* logRs  = (float*)alloc(BN * KB * 4);
  float* logRss = (float*)alloc(BN * KB * KB * 4);
  float* ent_plot   = (float*)alloc(BN * 4);
  float* entropy_el = (float*)alloc(BN * 4);

  // ---- state init (every call: deterministic, graph-safe) ----
  hipMemsetAsync(h_cur, 0, (size_t)RR * HD * 2, stream);
  hipMemsetAsync(c_cur, 0, (size_t)RR * HD * 4, stream);
  hipMemsetAsync(samples, 0, (size_t)RR * LP * 4, stream);
  hipMemsetAsync(outputs, 0, (size_t)RR * LP * 4, stream);
  hipMemsetAsync(logp, 0, RR * 4, stream);
  hipMemsetAsync(Gg, 0, RR * 4, stream);
  hipMemsetAsync(ent, 0, RR * 4, stream);
  hipMemsetAsync(ent_plot, 0, BN * 4, stream);
  hipMemsetAsync(entropy_el, 0, BN * 4, stream);
  init_sid_kernel<<<(RR + 255) / 256, 256, 0, stream>>>(sid);

  // ---- precompute: bf16 weight copies ----
  auto conv = [&](const float* s, bf16* d, int n) {
    f2b_kernel<<<(n + 255) / 256, 256, 0, stream>>>(s, d, n);
  };
  conv(W_hh, Whh_b, NG * HD);
  conv((const float*)d_in[7], Wfc1_b, HD * HD);
  conv(W_out, Wout_b, NV * HD);
  conv(x_feat, xfeat_b, BN * INSZ);
  f2b_slice_kernel<<<(NG * INSZ + 255) / 256, 256, 0, stream>>>(W_ih, Wih0_b, NG, INSZ, INSZ + OPSZ, 0);
  f2b_slice_kernel<<<(NG * OPSZ + 255) / 256, 256, 0, stream>>>(W_ih, Wihop_b, NG, OPSZ, INSZ + OPSZ, INSZ);
  inpvec_kernel<<<(416 * OPSZ + 255) / 256, 256, 0, stream>>>(W_in_op, b_in_op, inpvec_b);
  bsum_kernel<<<(NG + 255) / 256, 256, 0, stream>>>(b_ih, b_hh, bsum);

  // xc64 = x_feat @ W_ih[:, :1024]^T  (M=64 -> 4 Mtiles, N=4096, NT=4)
  {
    int waves = 4 * (NG / 64);
    gemm_bf16_kernel<4, INSZ, INSZ, INSZ, false, true, false>
        <<<(waves * 32 + 255) / 256, 256, 0, stream>>>(
        xfeat_b, Wih0_b, nullptr, xc64, nullptr, 4, NG);
  }
  // table = inp_vec @ W_ih[:,1024:]^T + (b_ih+b_hh)  (M=416 -> 26 Mtiles)
  {
    int waves = 26 * (NG / 64);
    gemm_bf16_kernel<4, OPSZ, OPSZ, OPSZ, false, true, false>
        <<<(waves * 32 + 255) / 256, 256, 0, stream>>>(
        inpvec_b, Wihop_b, bsum, table, nullptr, 26, NG);
  }

  // ---- 13 recurrent steps ----
  for (int t = 0; t < LP; ++t) {
    // fused gates GEMM + LSTM cell: h_cur,c_cur -> h_tmp,c_tmp
    gates_lstm_kernel<<<256, 256, 0, stream>>>(h_cur, Whh_b, xc64, table, sid,
                                               c_cur, c_tmp, h_tmp);
    // hd = relu(h @ W_fc1^T + b_fc1)
    {
      int waves = 32 * (HD / 64);
      gemm_bf16_kernel<4, HD, HD, HD, true, false, true>
          <<<(waves * 32 + 255) / 256, 256, 0, stream>>>(
          h_tmp, Wfc1_b, b_fc1, nullptr, hd_b, 32, HD);
    }
    // logits = hd @ W_out^T + b_out  (N=400 -> 5 groups of 80, NT=5)
    {
      int waves = 32 * (NV / 80);
      gemm_bf16_kernel<5, HD, HD, HD, false, true, false>
          <<<(waves * 32 + 255) / 256, 256, 0, stream>>>(
          hd_b, Wout_b, b_out, logits, nullptr, 32, NV);
    }
    // per-row softmax / entropy / Gumbel-with-max
    row_softmax_gumbel<<<RR, 128, 0, stream>>>(logits, logp, Gg,
                                               gumbel + (size_t)t * RR * NV,
                                               lpmat, gmat, ent_pre);
    // per-batch top-k + state reorder (samples/outputs/logp/G/ent/sid/order/phi_k)
    select_topk<<<BN, 256, 0, stream>>>(gmat, lpmat, logp, Gg, ent, ent_pre,
                                        samples, outputs, sid, order_r, phi_k, t);
    // gather h,c by parent row
    reorder_hc<<<(RR * HD) / 256, 256, 0, stream>>>(h_tmp, c_tmp, order_r, h_cur, c_cur);
    // compute_log_R
    logR_points<<<(BN * 1000 + 255) / 256, 256, 0, stream>>>(phi_k, Tbuf, basebuf);
    logR_reduce<<<BN, 256, 0, stream>>>(Tbuf, basebuf, phi_k, ent,
                                        logRs, logRss, ent_plot, entropy_el);
  }

  // ---- pack outputs ----
  int total = 18560;
  pack_out<<<(total + 255) / 256, 256, 0, stream>>>(outputs, samples, entropy_el,
                                                    logRs, logRss, phi_k, ent_plot,
                                                    (float*)d_out, total);
}
